// GraphTransformerLayer_58480274702581
// MI455X (gfx1250) — compile-verified, matrix operand order checked by reading, not run on hardware
//
#include <hip/hip_runtime.h>
#include <hip/hip_bf16.h>
#include <math.h>

typedef __attribute__((ext_vector_type(2)))  float   v2f;
typedef __attribute__((ext_vector_type(8)))  float   v8f;
typedef __attribute__((ext_vector_type(16))) __bf16  v16bf;
typedef unsigned short u16;
typedef unsigned char  u8;
typedef unsigned int   u32;

#define NN   2048
#define DD   512
#define HH   8
#define DHH  64
#define FFD  2048
#define BF16_ONE 0x3F80u

// ---------------------------------------------------------------------------
// CDNA5 async global->LDS copy (16B per lane), tracked by ASYNCcnt.
// GVS mode: mem = SGPR base + 32-bit lane offset; LDS dest = lane VGPR (byte).
// Low 32 bits of a generic shared pointer are the LDS byte address (aperture).
// ---------------------------------------------------------------------------
__device__ __forceinline__ void async_ld128(const void* sbase, unsigned goff, unsigned ldsb) {
    asm volatile("global_load_async_to_lds_b128 %0, %1, %2"
                 :: "v"(ldsb), "v"(goff), "s"(sbase) : "memory");
}
__device__ __forceinline__ void async_wait0() {
    asm volatile("s_wait_asynccnt 0" ::: "memory");
}
__device__ __forceinline__ unsigned lds_addr(const void* p) {
    return (unsigned)(uintptr_t)p;
}

// ---------------------------------------------------------------------------
// build G = I | (adj>0) as bf16 0/1 (plus its transpose), S = reach_0+reach_1
// ---------------------------------------------------------------------------
__global__ __launch_bounds__(256) void build_g(const float* __restrict__ adj,
                                               u16* __restrict__ G,
                                               u16* __restrict__ GT,
                                               u8* __restrict__ S) {
    size_t idx = (size_t)blockIdx.x * 256 + threadIdx.x;      // 4M elements
    int i = (int)(idx >> 11);
    int j = (int)(idx & 2047);
    bool a  = adj[idx] > 0.0f;
    bool dg = (i == j);
    bool r1 = a || dg;
    u16 val = r1 ? (u16)BF16_ONE : (u16)0;
    G[idx] = val;
    GT[(size_t)j * NN + i] = val;
    S[idx] = (u8)((dg ? 1 : 0) + (r1 ? 1 : 0));
}

// ---------------------------------------------------------------------------
// Reachability GEMM: Rn = binarize(Rp @ G), S += Rn   (bf16 WMMA, exact)
// 128x128 / WG, 4 waves 2x2, Kc=32, async double-buffered LDS staging.
// B operand fed from GT so both stagings are contiguous row copies.
// ---------------------------------------------------------------------------
#define RPAD 40  // u16 per LDS row: 80B = 16B aligned, conflict-free (20m mod 64)
__global__ __launch_bounds__(128) void reach_gemm(const u16* __restrict__ Rp,
                                                  const u16* __restrict__ GT,
                                                  u16* __restrict__ Rn,
                                                  u8* __restrict__ S) {
    __shared__ u16 Al[2][128 * RPAD];
    __shared__ u16 Bl[2][128 * RPAD];
    const int tid  = threadIdx.x;
    const int wid  = tid >> 5;
    const int lane = tid & 31;
    const int lm   = lane & 15;
    const int half = lane >> 4;
    const int wm   = (wid >> 1) * 64;
    const int wn   = (wid & 1) * 64;
    const int m0   = blockIdx.x * 128;
    const int n0   = blockIdx.y * 128;

    const unsigned ga0 = (unsigned)((size_t)(m0 + tid) * NN * 2);
    const unsigned gb0 = (unsigned)((size_t)(n0 + tid) * NN * 2);

    v8f acc[4][4];
#pragma unroll
    for (int i = 0; i < 4; ++i)
#pragma unroll
        for (int j = 0; j < 4; ++j)
#pragma unroll
            for (int e = 0; e < 8; ++e) acc[i][j][e] = 0.0f;

    union U16V { v16bf v; u32 u[8]; };

    // prologue: stage tile 0 into buffer 0
    {
        unsigned la = lds_addr(&Al[0][tid * RPAD]);
        unsigned lb = lds_addr(&Bl[0][tid * RPAD]);
#pragma unroll
        for (int i = 0; i < 4; ++i) {
            async_ld128(Rp, ga0 + i * 16, la + i * 16);
            async_ld128(GT, gb0 + i * 16, lb + i * 16);
        }
    }

    int buf = 0;
    for (int k0 = 0; k0 < NN; k0 += 32) {
        async_wait0();
        __syncthreads();
        if (k0 + 32 < NN) {
            unsigned la = lds_addr(&Al[buf ^ 1][tid * RPAD]);
            unsigned lb = lds_addr(&Bl[buf ^ 1][tid * RPAD]);
            unsigned ka = ga0 + (unsigned)(k0 + 32) * 2;
            unsigned kb = gb0 + (unsigned)(k0 + 32) * 2;
#pragma unroll
            for (int i = 0; i < 4; ++i) {
                async_ld128(Rp, ka + i * 16, la + i * 16);
                async_ld128(GT, kb + i * 16, lb + i * 16);
            }
        }

        U16V a[4], b[4];
#pragma unroll
        for (int t = 0; t < 4; ++t) {
            const char* ar = (const char*)&Al[buf][(wm + t * 16 + lm) * RPAD];
            const char* br = (const char*)&Bl[buf][(wn + t * 16 + lm) * RPAD];
#pragma unroll
            for (int p = 0; p < 8; ++p) {
                const int k = (p & 3) * 2 + (p >> 2) * 16 + half * 8;
                a[t].u[p] = *(const u32*)(ar + k * 2);
                b[t].u[p] = *(const u32*)(br + k * 2);
            }
        }
#pragma unroll
        for (int mt = 0; mt < 4; ++mt)
#pragma unroll
            for (int nt = 0; nt < 4; ++nt)
                acc[mt][nt] = __builtin_amdgcn_wmma_f32_16x16x32_bf16(
                    false, a[mt].v, false, b[nt].v, (short)0, acc[mt][nt], false, false);
        buf ^= 1;
    }

#pragma unroll
    for (int mt = 0; mt < 4; ++mt)
#pragma unroll
        for (int nt = 0; nt < 4; ++nt) {
            const int col = n0 + wn + nt * 16 + lm;
#pragma unroll
            for (int v = 0; v < 8; ++v) {
                const int row = m0 + wm + mt * 16 + v + 8 * half;
                const bool r  = acc[mt][nt][v] > 0.5f;
                const size_t idx = (size_t)row * NN + col;
                Rn[idx] = r ? (u16)BF16_ONE : (u16)0;
                S[idx] = (u8)(S[idx] + (r ? 1 : 0));
            }
        }
}

// ---------------------------------------------------------------------------
// bias[i][j] = spatial_emb[10 - S[i][j]]
// ---------------------------------------------------------------------------
__global__ __launch_bounds__(256) void bias_from_s(const u8* __restrict__ S,
                                                   const float* __restrict__ emb,
                                                   float* __restrict__ bias) {
    size_t idx = (size_t)blockIdx.x * 256 + threadIdx.x;
    int d = 10 - (int)S[idx];
    if (d < 0) d = 0;
    if (d > 10) d = 10;
    bias[idx] = emb[d];
}

// ---------------------------------------------------------------------------
// Generic NT fp32 GEMM on f32 WMMA:  C = act(A[M,K] @ B[N,K]^T + bias) (+ res)
// 128x128 block, 4 waves 2x2, Kc=32, async double-buffered staging.
// ---------------------------------------------------------------------------
#define GPAD 36  // floats per LDS row (144B: 16B aligned, conflict-free)
__global__ __launch_bounds__(128) void gemm_nt_f32(const float* __restrict__ A,
                                                   const float* __restrict__ B,
                                                   const float* __restrict__ bias,
                                                   const float* __restrict__ res,
                                                   float* __restrict__ C,
                                                   int N, int Kd, int fuse_gelu) {
    __shared__ float Al[2][128 * GPAD];
    __shared__ float Bl[2][128 * GPAD];
    const int tid  = threadIdx.x;
    const int wid  = tid >> 5;
    const int lane = tid & 31;
    const int lm   = lane & 15;
    const int half = lane >> 4;
    const int kk   = half * 2;
    const int wm   = (wid >> 1) * 64;
    const int wn   = (wid & 1) * 64;
    const int m0   = blockIdx.x * 128;
    const int n0   = blockIdx.y * 128;

    const unsigned ga0 = (unsigned)((size_t)(m0 + tid) * Kd * 4);
    const unsigned gb0 = (unsigned)((size_t)(n0 + tid) * Kd * 4);

    v8f acc[4][4];
#pragma unroll
    for (int i = 0; i < 4; ++i)
#pragma unroll
        for (int j = 0; j < 4; ++j)
#pragma unroll
            for (int e = 0; e < 8; ++e) acc[i][j][e] = 0.0f;

    {
        unsigned la = lds_addr(&Al[0][tid * GPAD]);
        unsigned lb = lds_addr(&Bl[0][tid * GPAD]);
#pragma unroll
        for (int i = 0; i < 8; ++i) {
            async_ld128(A, ga0 + i * 16, la + i * 16);
            async_ld128(B, gb0 + i * 16, lb + i * 16);
        }
    }

    int buf = 0;
    for (int k0 = 0; k0 < Kd; k0 += 32) {
        async_wait0();
        __syncthreads();
        if (k0 + 32 < Kd) {
            unsigned la = lds_addr(&Al[buf ^ 1][tid * GPAD]);
            unsigned lb = lds_addr(&Bl[buf ^ 1][tid * GPAD]);
            unsigned ka = ga0 + (unsigned)(k0 + 32) * 4;
            unsigned kb = gb0 + (unsigned)(k0 + 32) * 4;
#pragma unroll
            for (int i = 0; i < 8; ++i) {
                async_ld128(A, ka + i * 16, la + i * 16);
                async_ld128(B, kb + i * 16, lb + i * 16);
            }
        }

#pragma unroll
        for (int kc = 0; kc < 32; kc += 4) {
            v2f af[4], bf[4];
#pragma unroll
            for (int t = 0; t < 4; ++t) {
                af[t] = *(const v2f*)(&Al[buf][(wm + t * 16 + lm) * GPAD + kc + kk]);
                bf[t] = *(const v2f*)(&Bl[buf][(wn + t * 16 + lm) * GPAD + kc + kk]);
            }
#pragma unroll
            for (int mt = 0; mt < 4; ++mt)
#pragma unroll
                for (int nt = 0; nt < 4; ++nt)
                    acc[mt][nt] = __builtin_amdgcn_wmma_f32_16x16x4_f32(
                        false, af[mt], false, bf[nt], (short)0, acc[mt][nt], false, false);
        }
        buf ^= 1;
    }

#pragma unroll
    for (int mt = 0; mt < 4; ++mt)
#pragma unroll
        for (int nt = 0; nt < 4; ++nt) {
            const int col = n0 + wn + nt * 16 + lm;
            const float bv = bias[col];
#pragma unroll
            for (int v = 0; v < 8; ++v) {
                const int row = m0 + wm + mt * 16 + v + 8 * half;
                float x = acc[mt][nt][v] + bv;
                if (fuse_gelu) x = 0.5f * x * (1.0f + erff(x * 0.70710678118654752f));
                if (res) x += res[(size_t)row * N + col];
                C[(size_t)row * N + col] = x;
            }
        }
}

// ---------------------------------------------------------------------------
// Fused attention (flash style): per (64 q-rows, head). 4 waves, 16 rows/wave.
// scores = Q K^T / 8 + bias ; online softmax ; ctx += P V  (all f32 WMMA)
// ---------------------------------------------------------------------------
__global__ __launch_bounds__(128) void attn_fused(const float* __restrict__ Q,
                                                  const float* __restrict__ Km,
                                                  const float* __restrict__ Vm,
                                                  const float* __restrict__ bias,
                                                  float* __restrict__ ctx) {
    __shared__ float Kl[64 * 68];       // [key][dh]
    __shared__ float Vl[64 * 68];       // [dh][key]  (transposed)
    __shared__ float Pl[4 * 16 * 66];   // per-wave P tile [row][key]
    const int tid  = threadIdx.x;
    const int wid  = tid >> 5;
    const int lane = tid & 31;
    const int lm   = lane & 15;
    const int half = lane >> 4;
    const int kk   = half * 2;
    const int h    = blockIdx.y;
    const int qrow = blockIdx.x * 64 + wid * 16;
    float* Pw = Pl + wid * 16 * 66;

    v2f qf[16];
#pragma unroll
    for (int c = 0; c < 16; ++c)
        qf[c] = *(const v2f*)(Q + (size_t)(qrow + lm) * DD + h * DHH + c * 4 + kk);

    v8f cacc[4];
#pragma unroll
    for (int dt = 0; dt < 4; ++dt)
#pragma unroll
        for (int e = 0; e < 8; ++e) cacc[dt][e] = 0.0f;
    float mst[8], lst[8];
#pragma unroll
    for (int v = 0; v < 8; ++v) { mst[v] = -1e30f; lst[v] = 0.0f; }

    const int krow = tid >> 1;
    const int koff = (tid & 1) * 32;
    const unsigned klds = lds_addr(Kl + (size_t)krow * 68 + koff);

    for (int j0 = 0; j0 < NN; j0 += 64) {
        // stage K block [64][64] via async DMA
        {
            unsigned gk = (unsigned)(((size_t)(j0 + krow) * DD + h * DHH + koff) * 4);
#pragma unroll
            for (int i = 0; i < 8; ++i) async_ld128(Km, gk + i * 16, klds + i * 16);
        }
        // stage V block transposed: Vl[dh][key]
#pragma unroll
        for (int p = 0; p < 8; ++p) {
            const int key = p * 8 + (tid >> 4);
            const int d4  = (tid & 15) * 4;
            float4 v4 = *(const float4*)(Vm + (size_t)(j0 + key) * DD + h * DHH + d4);
            Vl[(d4 + 0) * 68 + key] = v4.x;
            Vl[(d4 + 1) * 68 + key] = v4.y;
            Vl[(d4 + 2) * 68 + key] = v4.z;
            Vl[(d4 + 3) * 68 + key] = v4.w;
        }
        async_wait0();
        __syncthreads();

        // scores: 4 tiles of 16 keys
        v8f s[4];
#pragma unroll
        for (int nt = 0; nt < 4; ++nt) {
#pragma unroll
            for (int e = 0; e < 8; ++e) s[nt][e] = 0.0f;
#pragma unroll
            for (int c = 0; c < 16; ++c) {
                v2f bf = *(const v2f*)(Kl + (size_t)(nt * 16 + lm) * 68 + c * 4 + kk);
                s[nt] = __builtin_amdgcn_wmma_f32_16x16x4_f32(
                    false, qf[c], false, bf, (short)0, s[nt], false, false);
            }
        }
        // scale + spatial bias
#pragma unroll
        for (int nt = 0; nt < 4; ++nt) {
            const int col = j0 + nt * 16 + lm;
#pragma unroll
            for (int v = 0; v < 8; ++v) {
                const int row = qrow + v + 8 * half;
                s[nt][v] = s[nt][v] * 0.125f + bias[(size_t)row * NN + col];
            }
        }
        // online softmax update
        float alp[8];
#pragma unroll
        for (int v = 0; v < 8; ++v) {
            float mx = fmaxf(fmaxf(s[0][v], s[1][v]), fmaxf(s[2][v], s[3][v]));
            mx = fmaxf(mx, __shfl_xor(mx, 1, 32));
            mx = fmaxf(mx, __shfl_xor(mx, 2, 32));
            mx = fmaxf(mx, __shfl_xor(mx, 4, 32));
            mx = fmaxf(mx, __shfl_xor(mx, 8, 32));
            const float mo = mst[v];
            const float mn = fmaxf(mo, mx);
            alp[v] = __expf(mo - mn);
            mst[v] = mn;
        }
#pragma unroll
        for (int nt = 0; nt < 4; ++nt)
#pragma unroll
            for (int v = 0; v < 8; ++v)
                s[nt][v] = __expf(s[nt][v] - mst[v]);
#pragma unroll
        for (int v = 0; v < 8; ++v) {
            float rs = s[0][v] + s[1][v] + s[2][v] + s[3][v];
            rs += __shfl_xor(rs, 1, 32);
            rs += __shfl_xor(rs, 2, 32);
            rs += __shfl_xor(rs, 4, 32);
            rs += __shfl_xor(rs, 8, 32);
            lst[v] = lst[v] * alp[v] + rs;
        }
#pragma unroll
        for (int dt = 0; dt < 4; ++dt)
#pragma unroll
            for (int v = 0; v < 8; ++v) cacc[dt][v] *= alp[v];
        // stage P tile (C-layout -> LDS row-major) for A-fragment reads
#pragma unroll
        for (int nt = 0; nt < 4; ++nt)
#pragma unroll
            for (int v = 0; v < 8; ++v)
                Pw[(v + 8 * half) * 66 + nt * 16 + lm] = s[nt][v];
        __syncthreads();

        // ctx += P @ V
#pragma unroll
        for (int dt = 0; dt < 4; ++dt) {
#pragma unroll
            for (int c = 0; c < 16; ++c) {
                v2f af = *(const v2f*)(Pw + (size_t)lm * 66 + c * 4 + kk);
                v2f bf = *(const v2f*)(Vl + (size_t)(dt * 16 + lm) * 68 + c * 4 + kk);
                cacc[dt] = __builtin_amdgcn_wmma_f32_16x16x4_f32(
                    false, af, false, bf, (short)0, cacc[dt], false, false);
            }
        }
        __syncthreads();
    }

#pragma unroll
    for (int dt = 0; dt < 4; ++dt)
#pragma unroll
        for (int v = 0; v < 8; ++v) {
            const int row = qrow + v + 8 * half;
            ctx[(size_t)row * DD + h * DHH + dt * 16 + lm] = cacc[dt][v] / lst[v];
        }
}

// ---------------------------------------------------------------------------
// LayerNorm over rows of 512
// ---------------------------------------------------------------------------
__global__ __launch_bounds__(128) void ln_kernel(const float* __restrict__ X,
                                                 const float* __restrict__ w,
                                                 const float* __restrict__ b,
                                                 float* __restrict__ Y) {
    __shared__ float red[128];
    const int row = blockIdx.x;
    const int tid = threadIdx.x;
    const float4 xv = ((const float4*)(X + (size_t)row * DD))[tid];

    red[tid] = xv.x + xv.y + xv.z + xv.w;
    __syncthreads();
    for (int o = 64; o > 0; o >>= 1) {
        if (tid < o) red[tid] += red[tid + o];
        __syncthreads();
    }
    const float mu = red[0] * (1.0f / DD);
    __syncthreads();

    float dx = xv.x - mu, dy = xv.y - mu, dz = xv.z - mu, dw = xv.w - mu;
    red[tid] = dx * dx + dy * dy + dz * dz + dw * dw;
    __syncthreads();
    for (int o = 64; o > 0; o >>= 1) {
        if (tid < o) red[tid] += red[tid + o];
        __syncthreads();
    }
    const float rstd = rsqrtf(red[0] * (1.0f / DD) + 1e-5f);
    __syncthreads();

    const int c = tid * 4;
    float* yr = Y + (size_t)row * DD;
    yr[c + 0] = dx * rstd * w[c + 0] + b[c + 0];
    yr[c + 1] = dy * rstd * w[c + 1] + b[c + 1];
    yr[c + 2] = dz * rstd * w[c + 2] + b[c + 2];
    yr[c + 3] = dw * rstd * w[c + 3] + b[c + 3];
}

// ---------------------------------------------------------------------------
extern "C" void kernel_launch(void* const* d_in, const int* in_sizes, int n_in,
                              void* d_out, int out_size, void* d_ws, size_t ws_size,
                              hipStream_t stream) {
    (void)in_sizes; (void)n_in; (void)out_size; (void)ws_size;
    const float* x    = (const float*)d_in[0];
    const float* adj  = (const float*)d_in[1];
    const float* emb  = (const float*)d_in[2];
    const float* wq   = (const float*)d_in[3];
    const float* bq   = (const float*)d_in[4];
    const float* wk   = (const float*)d_in[5];
    const float* bk   = (const float*)d_in[6];
    const float* wv   = (const float*)d_in[7];
    const float* bv   = (const float*)d_in[8];
    const float* wo   = (const float*)d_in[9];
    const float* bo   = (const float*)d_in[10];
    const float* ln1w = (const float*)d_in[11];
    const float* ln1b = (const float*)d_in[12];
    const float* w1   = (const float*)d_in[13];
    const float* b1   = (const float*)d_in[14];
    const float* w2   = (const float*)d_in[15];
    const float* b2   = (const float*)d_in[16];
    const float* ln2w = (const float*)d_in[17];
    const float* ln2b = (const float*)d_in[18];
    float* out = (float*)d_out;

    char* ws = (char*)d_ws;
    size_t off = 0;
    auto alloc = [&](size_t bytes) {
        char* p = ws + off;
        off += (bytes + 255) & ~(size_t)255;
        return p;
    };
    u16*   G    = (u16*)alloc((size_t)NN * NN * 2);
    u16*   GTr  = (u16*)alloc((size_t)NN * NN * 2);
    u16*   R0   = (u16*)alloc((size_t)NN * NN * 2);
    u16*   R1   = (u16*)alloc((size_t)NN * NN * 2);
    u8*    Sb   = (u8*)alloc((size_t)NN * NN);
    float* bias = (float*)alloc((size_t)NN * NN * 4);
    float* Qb   = (float*)alloc((size_t)NN * DD * 4);
    float* Kb   = (float*)alloc((size_t)NN * DD * 4);
    float* Vb   = (float*)alloc((size_t)NN * DD * 4);
    float* ctx  = (float*)alloc((size_t)NN * DD * 4);
    float* res1 = (float*)alloc((size_t)NN * DD * 4);
    float* hbuf = (float*)alloc((size_t)NN * DD * 4);
    float* mid  = (float*)alloc((size_t)NN * FFD * 4);
    float* res2 = (float*)alloc((size_t)NN * DD * 4);

    const int elem_blocks = (NN * NN) / 256;

    // 1) G, G^T and S init
    build_g<<<elem_blocks, 256, 0, stream>>>(adj, G, GTr, Sb);

    // 2) reach_t for t=2..9 (8 exact bf16 matmuls), accumulating S
    {
        const u16* src = G;
        u16* bufs[2] = {R0, R1};
        for (int t = 0; t < 8; ++t) {
            u16* dst = bufs[t & 1];
            reach_gemm<<<dim3(16, 16), 128, 0, stream>>>(src, GTr, dst, Sb);
            src = dst;
        }
    }

    // 3) spatial bias
    bias_from_s<<<elem_blocks, 256, 0, stream>>>(Sb, emb, bias);

    // 4) Q, K, V projections (fp32 WMMA)
    gemm_nt_f32<<<dim3(16, 4), 128, 0, stream>>>(x, wq, bq, nullptr, Qb, DD, DD, 0);
    gemm_nt_f32<<<dim3(16, 4), 128, 0, stream>>>(x, wk, bk, nullptr, Kb, DD, DD, 0);
    gemm_nt_f32<<<dim3(16, 4), 128, 0, stream>>>(x, wv, bv, nullptr, Vb, DD, DD, 0);

    // 5) fused attention
    attn_fused<<<dim3(NN / 64, HH), 128, 0, stream>>>(Qb, Kb, Vb, bias, ctx);

    // 6) output projection + residual(x)
    gemm_nt_f32<<<dim3(16, 4), 128, 0, stream>>>(ctx, wo, bo, x, res1, DD, DD, 0);

    // 7) LN1
    ln_kernel<<<NN, 128, 0, stream>>>(res1, ln1w, ln1b, hbuf);

    // 8) FFN1 with exact GELU
    gemm_nt_f32<<<dim3(16, 16), 128, 0, stream>>>(hbuf, w1, b1, nullptr, mid, FFD, DD, 1);

    // 9) FFN2 + residual(h)
    gemm_nt_f32<<<dim3(16, 4), 128, 0, stream>>>(mid, w2, b2, hbuf, res2, DD, FFD, 0);

    // 10) LN2 -> output
    ln_kernel<<<NN, 128, 0, stream>>>(res2, ln2w, ln2b, out);
}
